// Net_34729105555676
// MI455X (gfx1250) — 8-bit matrix operand order repaired and checked by reading, compiled, not run on hardware
//
#include <hip/hip_runtime.h>

typedef __attribute__((ext_vector_type(16))) int   v16i;
typedef __attribute__((ext_vector_type(8)))  float v8f;

#define FP8_ONE 0x38

#define ASYNC_COPY_B128(lds_off, gptr)                                         \
    asm volatile("global_load_async_to_lds_b128 %0, %1, off" ::"v"(lds_off),   \
                 "v"(gptr)                                                     \
                 : "memory")

#define WAIT_ASYNCCNT_0() asm volatile("s_wait_asynccnt 0x0" ::: "memory")

__global__ __launch_bounds__(256)
void k_binarize_fp8(const float* __restrict__ in, unsigned char* __restrict__ out,
                    long n) {
    long idx = ((long)blockIdx.x * blockDim.x + threadIdx.x) * 4;
    if (idx + 3 < n) {
        float4 v = *reinterpret_cast<const float4*>(in + idx);
        uchar4 o;
        o.x = (v.x >= 0.5f) ? (unsigned char)FP8_ONE : (unsigned char)0;
        o.y = (v.y >= 0.5f) ? (unsigned char)FP8_ONE : (unsigned char)0;
        o.z = (v.z >= 0.5f) ? (unsigned char)FP8_ONE : (unsigned char)0;
        o.w = (v.w >= 0.5f) ? (unsigned char)FP8_ONE : (unsigned char)0;
        *reinterpret_cast<uchar4*>(out + idx) = o;
    }
}

__global__ __launch_bounds__(256)
void k_transpose_b8(const unsigned char* __restrict__ src,
                    unsigned char* __restrict__ dst, int rows, int cols) {
    __shared__ unsigned char tile[32][33];
    const int c0 = blockIdx.x * 32, r0 = blockIdx.y * 32;
    const int tx = threadIdx.x, ty = threadIdx.y;
#pragma unroll
    for (int j = 0; j < 32; j += 8)
        tile[ty + j][tx] = src[(size_t)(r0 + ty + j) * cols + (c0 + tx)];
    __syncthreads();
#pragma unroll
    for (int j = 0; j < 32; j += 8)
        dst[(size_t)(c0 + ty + j) * rows + (r0 + tx)] = tile[tx][ty + j];
}

template <bool STORE_FP8, bool TWO_BIAS>
__global__ __launch_bounds__(256)
void k_gemm_bin(const unsigned char* __restrict__ A,
                const unsigned char* __restrict__ Bm,
                const float* __restrict__ bias0,
                const float* __restrict__ bias1,
                void* __restrict__ Cout,
                int M, int N, int K) {
    constexpr int LSTR = 144;
    __shared__ unsigned char lA[2][128 * LSTR];
    __shared__ unsigned char lB[2][128 * LSTR];

    const int tid  = threadIdx.x;
    const int lane = tid & 31;
    const int wave = tid >> 5;
    const int wm   = wave & 1;
    const int wn   = wave >> 1;
    const int l15  = lane & 15;
    const int lhi  = (lane >> 4) & 1;
    const int m_blk0 = blockIdx.y * 128;
    const int n_blk0 = blockIdx.x * 128;

    auto issue_stage = [&](int buf, int k0) {
#pragma unroll
        for (int t = 0; t < 4; ++t) {
            const int c   = tid + 256 * t;
            const int row = c >> 3;
            const int kk  = (c & 7) * 16;
            const unsigned aoff =
                (unsigned)(uintptr_t)&lA[buf][row * LSTR + kk];
            const unsigned char* ag = A + (size_t)(m_blk0 + row) * K + k0 + kk;
            ASYNC_COPY_B128(aoff, ag);
            const unsigned boff =
                (unsigned)(uintptr_t)&lB[buf][row * LSTR + kk];
            const unsigned char* bg = Bm + (size_t)(n_blk0 + row) * K + k0 + kk;
            ASYNC_COPY_B128(boff, bg);
        }
    };

    v8f acc[4][2];
#pragma unroll
    for (int i = 0; i < 4; ++i)
#pragma unroll
        for (int j = 0; j < 2; ++j)
            acc[i][j] = v8f{0.f, 0.f, 0.f, 0.f, 0.f, 0.f, 0.f, 0.f};

    const int nsteps = K >> 7;
    issue_stage(0, 0);

    for (int s = 0; s < nsteps; ++s) {
        const int buf = s & 1;
        WAIT_ASYNCCNT_0();
        __syncthreads();
        if (s + 1 < nsteps) issue_stage(buf ^ 1, (s + 1) << 7);

        union AF { v16i v; unsigned long long u[8]; } a[4];
#pragma unroll
        for (int mt = 0; mt < 4; ++mt) {
            const unsigned char* p =
                &lA[buf][(wm * 64 + mt * 16 + l15) * LSTR + lhi * 16];
#pragma unroll
            for (int c = 0; c < 8; ++c)
                a[mt].u[c] =
                    *reinterpret_cast<const unsigned long long*>(p + 32 * (c >> 1) + 8 * (c & 1));
        }
        union BF { v16i v; uint4 q[4]; } b[2];
#pragma unroll
        for (int nt = 0; nt < 2; ++nt) {
            const unsigned char* p =
                &lB[buf][(wn * 32 + nt * 16 + l15) * LSTR + lhi * 16];
#pragma unroll
            for (int c = 0; c < 4; ++c)
                b[nt].q[c] = *reinterpret_cast<const uint4*>(p + 32 * c);
        }
#pragma unroll
        for (int mt = 0; mt < 4; ++mt)
#pragma unroll
            for (int nt = 0; nt < 2; ++nt)
                acc[mt][nt] = __builtin_amdgcn_wmma_f32_16x16x128_fp8_fp8(
                    a[mt].v, b[nt].v, (short)0, acc[mt][nt], false, false);
    }

#pragma unroll
    for (int nt = 0; nt < 2; ++nt) {
        const int n = n_blk0 + wn * 32 + nt * 16 + l15;
        float bias = bias0[n];
        if (TWO_BIAS) bias += bias1[n];
#pragma unroll
        for (int mt = 0; mt < 4; ++mt) {
#pragma unroll
            for (int r = 0; r < 8; ++r) {
                const int m = m_blk0 + wm * 64 + mt * 16 + r + lhi * 8;
                const float v = acc[mt][nt][r] + bias;
                if (STORE_FP8) {
                    ((unsigned char*)Cout)[(size_t)m * N + n] =
                        (v >= 1.0f) ? (unsigned char)FP8_ONE : (unsigned char)0;
                } else {
                    ((float*)Cout)[(size_t)m * N + n] = (v >= 1.0f) ? 1.0f : 0.0f;
                }
            }
        }
    }
}

extern "C" void kernel_launch(void* const* d_in, const int* in_sizes, int n_in,
                              void* d_out, int out_size, void* d_ws, size_t ws_size,
                              hipStream_t stream) {
    const float* x     = (const float*)d_in[0];
    const float* W     = (const float*)d_in[1];
    const float* b_enc = (const float*)d_in[2];
    const float* b0    = (const float*)d_in[3];
    const float* b3    = (const float*)d_in[4];
    float* out = (float*)d_out;

    const int B = 8192, D = 4096, H = 2048;

    unsigned char* ws  = (unsigned char*)d_ws;
    unsigned char* Wb  = ws;
    unsigned char* WbT = Wb + (size_t)H * D;
    unsigned char* x8  = WbT + (size_t)D * H;
    unsigned char* h8  = x8 + (size_t)B * D;

    {
        long n = (long)H * D;
        k_binarize_fp8<<<(unsigned)(n / 1024), 256, 0, stream>>>(W, Wb, n);
    }
    {
        long n = (long)B * D;
        k_binarize_fp8<<<(unsigned)(n / 1024), 256, 0, stream>>>(x, x8, n);
    }
    k_transpose_b8<<<dim3(D / 32, H / 32), dim3(32, 8), 0, stream>>>(Wb, WbT, H, D);

    k_gemm_bin<true, true><<<dim3(H / 128, B / 128), 256, 0, stream>>>(
        x8, Wb, b_enc, b0, h8, B, H, D);

    k_gemm_bin<false, false><<<dim3(D / 128, B / 128), 256, 0, stream>>>(
        h8, WbT, b3, b3, out, B, D, H);
}
